// KMeans_44770739094236
// MI455X (gfx1250) — matrix operand order repaired and checked by reading, not compiled, not run on hardware
//
#include <hip/hip_runtime.h>

// ---------------- types for WMMA ----------------
typedef __attribute__((ext_vector_type(16))) __bf16        v16bf;
typedef __attribute__((ext_vector_type(8)))  float         v8f;
typedef __attribute__((ext_vector_type(16))) unsigned short v16us;
typedef __attribute__((ext_vector_type(8)))  unsigned short v8us;

#define NPTS      262144
#define KDIM      64
#define NCLUST    512
#define MAXITER   5

// round-to-nearest-even f32 -> bf16 (bit pattern)
__device__ __forceinline__ unsigned short f2bf(float f) {
    unsigned int u = __float_as_uint(f);
    u += 0x7FFFu + ((u >> 16) & 1u);
    return (unsigned short)(u >> 16);
}

// ---------------- one-time X -> bf16 ----------------
__global__ void convx_kernel(const float* __restrict__ X,
                             unsigned short* __restrict__ Xb, int nelem) {
    int i = (blockIdx.x * blockDim.x + threadIdx.x) * 4;
    if (i >= nelem) return;
    float4 v = *(const float4*)(X + i);
    ushort4 o;
    o.x = f2bf(v.x); o.y = f2bf(v.y); o.z = f2bf(v.z); o.w = f2bf(v.w);
    *(ushort4*)(Xb + i) = o;
}

// ---------------- per-iter: centroids f32 -> bf16 + |c|^2 ----------------
__global__ void prep_kernel(const float* __restrict__ cents,
                            unsigned short* __restrict__ Cb,
                            float* __restrict__ csq) {
    int c = blockIdx.x * blockDim.x + threadIdx.x;
    if (c >= NCLUST) return;
    const float* row = cents + c * KDIM;
    float s = 0.f;
#pragma unroll
    for (int d = 0; d < KDIM; ++d) {
        float v = row[d];
        s += v * v;
        Cb[c * KDIM + d] = f2bf(v);
    }
    csq[c] = s;
}

// ---------------- assignment: WMMA bf16 GEMM + argmax ----------------
// block = 256 threads = 8 waves; each wave: 2 A-tiles (32 rows) vs all 512
// clusters; B tile software-pipelined so loads overlap the 4 WMMAs.
__global__ void __launch_bounds__(256)
assign_kernel(const unsigned short* __restrict__ Xb,
              const unsigned short* __restrict__ Cb,
              const float* __restrict__ csq,
              int* __restrict__ closest) {
    const int lane     = threadIdx.x & 31;
    const int wave     = threadIdx.x >> 5;
    const int row_base = blockIdx.x * 256 + wave * 32;
    const int half     = lane >> 4;   // 0: lanes 0-15, 1: lanes 16-31
    const int l16      = lane & 15;

    // ---- A tiles: 2 x (16 rows x K=64), contiguous K layout (k = 16*half + s) ----
    // lanes 0-15: K in {0..15} (+32 for 2nd); lanes 16-31: {16..31}
    const int ka = half * 16;
    const unsigned short* xrow0 = Xb + (size_t)(row_base + l16) * KDIM;
    const unsigned short* xrow1 = Xb + (size_t)(row_base + 16 + l16) * KDIM;
    v16us a0u, a1u, a2u, a3u;
    ((v8us*)&a0u)[0] = *(const v8us*)(xrow0 + ka);
    ((v8us*)&a0u)[1] = *(const v8us*)(xrow0 + ka + 8);
    ((v8us*)&a1u)[0] = *(const v8us*)(xrow0 + ka + 32);
    ((v8us*)&a1u)[1] = *(const v8us*)(xrow0 + ka + 40);
    ((v8us*)&a2u)[0] = *(const v8us*)(xrow1 + ka);
    ((v8us*)&a2u)[1] = *(const v8us*)(xrow1 + ka + 8);
    ((v8us*)&a3u)[0] = *(const v8us*)(xrow1 + ka + 32);
    ((v8us*)&a3u)[1] = *(const v8us*)(xrow1 + ka + 40);
    const v16bf a0 = __builtin_bit_cast(v16bf, a0u);
    const v16bf a1 = __builtin_bit_cast(v16bf, a1u);
    const v16bf a2 = __builtin_bit_cast(v16bf, a2u);
    const v16bf a3 = __builtin_bit_cast(v16bf, a3u);

    float best0[8], best1[8];
    int   bidx0[8], bidx1[8];
#pragma unroll
    for (int i = 0; i < 8; ++i) {
        best0[i] = -3.4e38f; bidx0[i] = 0;
        best1[i] = -3.4e38f; bidx1[i] = 0;
    }

    // ---- software-pipelined sweep of 32 centroid tiles of 16 ----
    // B tile: K=32 x N=16; lane holds N=l16, one contiguous 32B run of K
    const int kb = half * 16;
    const unsigned short* crow = Cb + (size_t)l16 * KDIM;
    v16us nb0 = *(const v16us*)(crow + kb);
    v16us nb1 = *(const v16us*)(crow + kb + 32);
    float ncs = csq[l16];

    for (int t = 0; t < 32; ++t) {
        const v16bf b0 = __builtin_bit_cast(v16bf, nb0);
        const v16bf b1 = __builtin_bit_cast(v16bf, nb1);
        const float cs = ncs;

        // prefetch next tile (wrapped; extra wrap load is harmless)
        const int tn = (t + 1) & 31;
        const unsigned short* crn = Cb + (size_t)(tn * 16 + l16) * KDIM;
        nb0 = *(const v16us*)(crn + kb);
        nb1 = *(const v16us*)(crn + kb + 32);
        ncs = csq[tn * 16 + l16];

        v8f c0 = {}, c1 = {};
        c0 = __builtin_amdgcn_wmma_f32_16x16x32_bf16(false, a0, false, b0,
                                                     (short)0, c0, false, false);
        c1 = __builtin_amdgcn_wmma_f32_16x16x32_bf16(false, a2, false, b0,
                                                     (short)0, c1, false, false);
        c0 = __builtin_amdgcn_wmma_f32_16x16x32_bf16(false, a1, false, b1,
                                                     (short)0, c0, false, false);
        c1 = __builtin_amdgcn_wmma_f32_16x16x32_bf16(false, a3, false, b1,
                                                     (short)0, c1, false, false);

        const int cluster = t * 16 + l16;
#pragma unroll
        for (int i = 0; i < 8; ++i) {
            float s0 = 2.0f * c0[i] - cs;
            if (s0 > best0[i]) { best0[i] = s0; bidx0[i] = cluster; }
            float s1 = 2.0f * c1[i] - cs;
            if (s1 > best1[i]) { best1[i] = s1; bidx1[i] = cluster; }
        }
    }

    // ---- argmax across the 16 lanes of each half (D: M=i+8*half, N=l16) ----
#pragma unroll
    for (int off = 1; off < 16; off <<= 1) {
#pragma unroll
        for (int i = 0; i < 8; ++i) {
            float ov = __shfl_xor(best0[i], off, 32);
            int   oi = __shfl_xor(bidx0[i], off, 32);
            if (ov > best0[i] || (ov == best0[i] && oi < bidx0[i])) {
                best0[i] = ov; bidx0[i] = oi;
            }
            ov = __shfl_xor(best1[i], off, 32);
            oi = __shfl_xor(bidx1[i], off, 32);
            if (ov > best1[i] || (ov == best1[i] && oi < bidx1[i])) {
                best1[i] = ov; bidx1[i] = oi;
            }
        }
    }
    if (l16 == 0) {
#pragma unroll
        for (int i = 0; i < 8; ++i) {
            closest[row_base + half * 8 + i]      = bidx0[i];
            closest[row_base + 16 + half * 8 + i] = bidx1[i];
        }
    }
}

// ---------------- scatter-mean accumulation via 130KB LDS histogram ----------------
__global__ void __launch_bounds__(256)
update_kernel(const float* __restrict__ X, const int* __restrict__ closest,
              float* __restrict__ gsums, float* __restrict__ gcounts,
              int pts_per_block) {
    __shared__ float ls[NCLUST * KDIM];   // 128 KB
    __shared__ float lc[NCLUST];          //   2 KB  (320KB/WGP on CDNA5)
    for (int i = threadIdx.x; i < NCLUST * KDIM; i += 256) ls[i] = 0.f;
    for (int i = threadIdx.x; i < NCLUST; i += 256) lc[i] = 0.f;
    __syncthreads();

    const int p0 = blockIdx.x * pts_per_block;
    for (int p = p0 + threadIdx.x; p < p0 + pts_per_block; p += 256) {
        const int c = closest[p];
        const float4* xr = (const float4*)(X + (size_t)p * KDIM);
        float* dst = ls + c * KDIM;
#pragma unroll
        for (int q = 0; q < KDIM / 4; ++q) {
            float4 v = xr[q];
            atomicAdd(dst + q * 4 + 0, v.x);
            atomicAdd(dst + q * 4 + 1, v.y);
            atomicAdd(dst + q * 4 + 2, v.z);
            atomicAdd(dst + q * 4 + 3, v.w);
        }
        atomicAdd(lc + c, 1.0f);
    }
    __syncthreads();
    for (int i = threadIdx.x; i < NCLUST * KDIM; i += 256) atomicAdd(gsums + i, ls[i]);
    for (int i = threadIdx.x; i < NCLUST; i += 256) atomicAdd(gcounts + i, lc[i]);
}

// ---------------- new centroids = sums / counts (empty -> 0) ----------------
__global__ void finalize_kernel(const float* __restrict__ sums,
                                const float* __restrict__ counts,
                                float* __restrict__ cents) {
    int idx = blockIdx.x * blockDim.x + threadIdx.x;
    if (idx >= NCLUST * KDIM) return;
    float cnt = counts[idx >> 6];
    cents[idx] = (cnt > 0.f) ? sums[idx] / cnt : 0.f;
}

// ---------------- final cost + emit labels ----------------
__global__ void cost_kernel(const float* __restrict__ X,
                            const float* __restrict__ cents,
                            const int* __restrict__ closest,
                            int* __restrict__ out_idx,
                            float* __restrict__ out_cost, int n) {
    int p = blockIdx.x * blockDim.x + threadIdx.x;
    float acc = 0.f;
    if (p < n) {
        int c = closest[p];
        out_idx[p] = c;
        const float* xr = X + (size_t)p * KDIM;
        const float* cr = cents + (size_t)c * KDIM;
#pragma unroll
        for (int d = 0; d < KDIM; ++d) {
            float df = cr[d] - xr[d];
            acc += df * df;
        }
    }
#pragma unroll
    for (int off = 16; off > 0; off >>= 1) acc += __shfl_down(acc, off, 32);
    if ((threadIdx.x & 31) == 0) atomicAdd(out_cost, acc);
}

extern "C" void kernel_launch(void* const* d_in, const int* in_sizes, int n_in,
                              void* d_out, int out_size, void* d_ws, size_t ws_size,
                              hipStream_t stream) {
    const float* X  = (const float*)d_in[0];   // [262144, 64] f32
    const float* C0 = (const float*)d_in[1];   // [512, 64] f32
    const int n = in_sizes[0] / KDIM;          // 262144

    // ---- workspace layout ----
    char* ws = (char*)d_ws;
    size_t off = 0;
    unsigned short* Xb   = (unsigned short*)(ws + off); off += (size_t)n * KDIM * 2;       // 32 MB
    float*          cents= (float*)         (ws + off); off += NCLUST * KDIM * 4;          // 128 KB
    unsigned short* Cb   = (unsigned short*)(ws + off); off += NCLUST * KDIM * 2;          // 64 KB
    float*          csq  = (float*)         (ws + off); off += NCLUST * 4;                 // 2 KB
    float*          sums = (float*)         (ws + off); off += NCLUST * KDIM * 4;          // 128 KB
    float*          cnts = (float*)         (ws + off); off += NCLUST * 4;                 // 2 KB (contiguous w/ sums)
    int*            clos = (int*)           (ws + off); off += (size_t)n * 4;              // 1 MB

    // centroids working copy (never mutate inputs)
    hipMemcpyAsync(cents, C0, NCLUST * KDIM * sizeof(float),
                   hipMemcpyDeviceToDevice, stream);

    // X -> bf16 once
    {
        int nelem = n * KDIM;
        int blocks = (nelem / 4 + 255) / 256;
        convx_kernel<<<blocks, 256, 0, stream>>>(X, Xb, nelem);
    }

    const int pts_per_block = n / 128;  // 2048
    for (int it = 0; it < MAXITER; ++it) {
        prep_kernel<<<(NCLUST + 255) / 256, 256, 0, stream>>>(cents, Cb, csq);
        assign_kernel<<<n / 256, 256, 0, stream>>>(Xb, Cb, csq, clos);
        hipMemsetAsync(sums, 0, (NCLUST * KDIM + NCLUST) * sizeof(float), stream);
        update_kernel<<<128, 256, 0, stream>>>(X, clos, sums, cnts, pts_per_block);
        finalize_kernel<<<(NCLUST * KDIM) / 256, 256, 0, stream>>>(sums, cnts, cents);
    }

    // d_out: [n] int32 labels, then 1 float cost
    int*   out_idx  = (int*)d_out;
    float* out_cost = (float*)((char*)d_out + (size_t)n * 4);
    hipMemsetAsync(out_cost, 0, sizeof(float), stream);
    cost_kernel<<<(n + 255) / 256, 256, 0, stream>>>(X, cents, clos, out_idx, out_cost, n);
}
